// KGHeteroGNN_49701361549687
// MI455X (gfx1250) — matrix operand order repaired and checked by reading, not compiled, not run on hardware
//
#include <hip/hip_runtime.h>
#include <cstdint>

typedef __bf16 bf16;
typedef __attribute__((ext_vector_type(16))) __bf16 v16bf;
typedef __attribute__((ext_vector_type(8)))  __bf16 v8bf;
typedef __attribute__((ext_vector_type(8)))  float  v8f;

#if __has_builtin(__builtin_amdgcn_global_load_async_to_lds_b128) && \
    __has_builtin(__builtin_amdgcn_s_wait_asynccnt)
#define USE_ASYNC_LDS 1
#define AS1 __attribute__((address_space(1)))
#define AS3 __attribute__((address_space(3)))
typedef int v4i_ __attribute__((__vector_size__(4 * sizeof(int))));
#endif

// ---------- helpers ----------
__device__ __forceinline__ unsigned short f2bf_bits(float f) {
    unsigned u = __float_as_uint(f);
    unsigned r = u + 0x7FFFu + ((u >> 16) & 1u);   // round-to-nearest-even
    return (unsigned short)(r >> 16);
}
__device__ __forceinline__ bf16 bfc(unsigned short h) {
    return __builtin_bit_cast(bf16, h);
}
__device__ __forceinline__ unsigned encOrd(float f) {   // order-preserving uint encoding
    unsigned u = __float_as_uint(f);
    return (u & 0x80000000u) ? ~u : (u | 0x80000000u);
}
__device__ __forceinline__ float decOrd(unsigned u) {
    unsigned b = (u & 0x80000000u) ? (u ^ 0x80000000u) : ~u;
    return __uint_as_float(b);
}

// ---------- fp32 -> bf16 conversion (activations) ----------
__global__ void cvt_bf16(const float* __restrict__ src, unsigned short* __restrict__ dst,
                         long long n) {
    long long i = (long long)blockIdx.x * blockDim.x + threadIdx.x;
    if (i < n) dst[i] = f2bf_bits(src[i]);
}

// ---------- weight pack: fp32 KxN row-major -> bf16 fragment order ----------
// packed[((k/32)*N + col)*32 + (k%32)] = bf16(W[k][col])
// so each lane's 16 B-fragment values (kbB + e) are 32 contiguous bytes.
__global__ void pack_b_bf16(const float* __restrict__ src, unsigned short* __restrict__ dst,
                            int K, int N) {
    int i = blockIdx.x * blockDim.x + threadIdx.x;
    if (i >= K * N) return;
    int k = i / N, c = i % N;
    dst[((size_t)(k >> 5) * N + c) * 32 + (k & 31)] = f2bf_bits(src[i]);
}

// ---------- WMMA bf16 GEMM: C[M,N] = A[M,K] @ Bpacked; A bf16 row-major, C fp32 ----------
// K % 32 == 0, N % 64 == 0. A staged in LDS (async b128); B read from global (L2-resident).
#define BM 128
#define BN 64
#define KC 32
#define LDA 40   // bf16 per A row in LDS: 80B (16B aligned)

__global__ __launch_bounds__(256) void gemm_wmma(const unsigned short* __restrict__ A,
                                                 const unsigned short* __restrict__ Bp,
                                                 float* __restrict__ C,
                                                 int M, int N, int K) {
    __shared__ __align__(16) unsigned short As[BM * LDA];
    const int tid  = threadIdx.x;
    const int lane = tid & 31;
    const int wave = tid >> 5;
    const int wm = wave & 3;          // 4 wave-rows of 32
    const int wn = wave >> 2;         // 2 wave-cols of 32
    const long long m0 = (long long)blockIdx.x * BM;
    const int n0 = blockIdx.y * BN;

    // A staging coords (branch-free: clamp tail rows; clamped rows feed unstored C rows)
    const int ra = tid >> 1;                 // 0..127
    const int ca = (tid & 1) * 16;           // 0 or 16
    long long arow_g = m0 + ra;
    if (arow_g >= M) arow_g = M - 1;
    const unsigned short* aSrc = A + arow_g * (long long)K;

    // fragment coords (contiguous k order k = 16*(lane/16) + e, same for A and B)
    const int fr   = lane & 15;
    const int kbA  = (lane >> 4) * 16;       // A: lanes 0-15 K 0-15; 16-31 K 16-31
    const int kbB  = (lane >> 4) * 16;       // B: lanes 0-15 K 0-15; 16-31 K 16-31
    const int a0r  = (wm * 32 + fr) * LDA;
    const int a1r  = (wm * 32 + 16 + fr) * LDA;

    // per-lane packed-B pointer: 16 contiguous bf16 per fragment
    const unsigned short* bPtr = Bp + ((size_t)(n0 + wn * 32 + fr)) * 32 + kbB;
    const size_t bStep = (size_t)N * 32;     // advance per k-step (32 K-rows)

    v8f acc00 = {}, acc01 = {}, acc10 = {}, acc11 = {};

    for (int k0 = 0; k0 < K; k0 += KC) {
#ifdef USE_ASYNC_LDS
        __builtin_amdgcn_global_load_async_to_lds_b128(
            (AS1 v4i_*)(aSrc + k0 + ca),     (AS3 v4i_*)(As + ra * LDA + ca), 0, 0);
        __builtin_amdgcn_global_load_async_to_lds_b128(
            (AS1 v4i_*)(aSrc + k0 + ca + 8), (AS3 v4i_*)(As + ra * LDA + ca + 8), 0, 0);
#else
        {
            const uint4 va0 = *(const uint4*)(aSrc + k0 + ca);
            const uint4 va1 = *(const uint4*)(aSrc + k0 + ca + 8);
            *(uint4*)&As[ra * LDA + ca]     = va0;
            *(uint4*)&As[ra * LDA + ca + 8] = va1;
        }
#endif
        // B fragments straight from global (packed, L2-resident weights)
        const v8bf bl0 = *(const v8bf*)(bPtr);
        const v8bf bh0 = *(const v8bf*)(bPtr + 8);
        const v8bf bl1 = *(const v8bf*)(bPtr + 512);        // +16 columns * 32
        const v8bf bh1 = *(const v8bf*)(bPtr + 520);
        const v16bf b0 = __builtin_shufflevector(bl0, bh0, 0,1,2,3,4,5,6,7,8,9,10,11,12,13,14,15);
        const v16bf b1 = __builtin_shufflevector(bl1, bh1, 0,1,2,3,4,5,6,7,8,9,10,11,12,13,14,15);
        bPtr += bStep;

#ifdef USE_ASYNC_LDS
        __builtin_amdgcn_s_wait_asynccnt(0);
#endif
        __syncthreads();

        v16bf af0, af1;
        #pragma unroll
        for (int e = 0; e < 8; ++e) {
            af0[e]     = bfc(As[a0r + kbA + e]);
            af0[e + 8] = bfc(As[a0r + kbA + 8 + e]);
            af1[e]     = bfc(As[a1r + kbA + e]);
            af1[e + 8] = bfc(As[a1r + kbA + 8 + e]);
        }
        acc00 = __builtin_amdgcn_wmma_f32_16x16x32_bf16(false, af0, false, b0, (short)0, acc00, false, false);
        acc01 = __builtin_amdgcn_wmma_f32_16x16x32_bf16(false, af0, false, b1, (short)0, acc01, false, false);
        acc10 = __builtin_amdgcn_wmma_f32_16x16x32_bf16(false, af1, false, b0, (short)0, acc10, false, false);
        acc11 = __builtin_amdgcn_wmma_f32_16x16x32_bf16(false, af1, false, b1, (short)0, acc11, false, false);
        __syncthreads();
    }

    // C/D layout: VGPR r -> M = r + 8*(lane>=16), N = lane&15
    const int csub = (lane >> 4) * 8;
    const long long c0 = n0 + wn * 32 + fr;
    const long long mBase = m0 + wm * 32 + csub;
    if (m0 + BM <= M) {
        // interior block: branch-free stores, incremental row pointer
        float* cp = C + mBase * (long long)N + c0;
        float* cq = cp + 16LL * N;
        #pragma unroll
        for (int r = 0; r < 8; ++r) {
            cp[0]  = acc00[r];
            cp[16] = acc01[r];
            cq[0]  = acc10[r];
            cq[16] = acc11[r];
            cp += N; cq += N;
        }
    } else {
        #pragma unroll
        for (int r = 0; r < 8; ++r) {
            long long mA = mBase + r;
            long long mB = mA + 16;
            if (mA < M) {
                C[mA * (long long)N + c0]      = acc00[r];
                C[mA * (long long)N + c0 + 16] = acc01[r];
            }
            if (mB < M) {
                C[mB * (long long)N + c0]      = acc10[r];
                C[mB * (long long)N + c0 + 16] = acc11[r];
            }
        }
    }
}

// ---------- fills ----------
__global__ void fill_f32(float* p, float v, long long n) {
    long long i = (long long)blockIdx.x * blockDim.x + threadIdx.x;
    if (i < n) p[i] = v;
}
__global__ void fill_u32(unsigned* p, unsigned v, int n) {
    int i = blockIdx.x * blockDim.x + threadIdx.x;
    if (i < n) p[i] = v;
}

// ---------- GATv2 edge passes (wave-per-edge) ----------
__global__ void edge_scores(const float* __restrict__ xl, const float* __restrict__ xr,
                            const float* __restrict__ att,
                            const int* __restrict__ esrc, const int* __restrict__ edst,
                            int E, int nsl, int H,
                            float* __restrict__ ebuf, unsigned* __restrict__ menc) {
    int wid  = (blockIdx.x * blockDim.x + threadIdx.x) >> 5;
    int lane = threadIdx.x & 31;
    if (wid >= E + nsl) return;
    int s, d;
    if (wid < E) { s = esrc[wid]; d = edst[wid]; } else { s = wid - E; d = s; }
    const float* rl = xl + (long long)s * H;
    const float* rr = xr + (long long)d * H;
    float acc = 0.f;
    for (int h = lane; h < H; h += 32) {
        float v = rl[h] + rr[h];
        v = (v > 0.f) ? v : 0.2f * v;     // leaky_relu(., 0.2)
        acc += v * att[h];
    }
    #pragma unroll
    for (int o = 16; o > 0; o >>= 1) acc += __shfl_xor(acc, o, 32);
    if (lane == 0) {
        ebuf[wid] = acc;
        atomicMax(&menc[d], encOrd(acc));
    }
}

__global__ void edge_softmax(const int* __restrict__ edst, int E, int nsl,
                             float* __restrict__ ebuf, const unsigned* __restrict__ menc,
                             float* __restrict__ den) {
    int i = blockIdx.x * blockDim.x + threadIdx.x;
    if (i >= E + nsl) return;
    int d = (i < E) ? edst[i] : (i - E);
    float w = __expf(ebuf[i] - decOrd(menc[d]));
    ebuf[i] = w;
    atomicAdd(&den[d], w);
}

__global__ void edge_aggregate(const float* __restrict__ xl, const float* __restrict__ ebuf,
                               const int* __restrict__ esrc, const int* __restrict__ edst,
                               int E, int nsl, int H, const float* __restrict__ den,
                               float* __restrict__ out) {
    int wid  = (blockIdx.x * blockDim.x + threadIdx.x) >> 5;
    int lane = threadIdx.x & 31;
    if (wid >= E + nsl) return;
    int s, d;
    if (wid < E) { s = esrc[wid]; d = edst[wid]; } else { s = wid - E; d = s; }
    float alpha = ebuf[wid] / den[d];
    const float* rl = xl + (long long)s * H;
    float* ro = out + (long long)d * H;
    for (int h = lane; h < H; h += 32)
        atomicAdd(&ro[h], alpha * rl[h]);
}

// ---------- epilogue ----------
__global__ void relu_bias(float* __restrict__ x, long long n, int H,
                          const float* __restrict__ b0, const float* __restrict__ b1,
                          const float* __restrict__ b2) {
    long long i = (long long)blockIdx.x * blockDim.x + threadIdx.x;
    if (i >= n) return;
    int h = (int)(i % H);
    float b = b0[h];
    if (b1) b += b1[h];
    if (b2) b += b2[h];
    float v = x[i] + b;
    x[i] = v > 0.f ? v : 0.f;
}

__global__ void final_linear(const float* __restrict__ x, const float* __restrict__ w,
                             const float* __restrict__ b, float* __restrict__ out,
                             int M, int K) {
    int i = blockIdx.x * blockDim.x + threadIdx.x;
    if (i >= M) return;
    float a0 = b[0], a1 = b[1];
    const float* row = x + (long long)i * K;
    for (int k = 0; k < K; ++k) {
        float v = row[k];
        a0 += v * w[2 * k];
        a1 += v * w[2 * k + 1];
    }
    out[2 * i]     = a0;
    out[2 * i + 1] = a1;
}

// ---------- host orchestration ----------
static inline int ceildiv(int a, int b) { return (a + b - 1) / b; }

extern "C" void kernel_launch(void* const* d_in, const int* in_sizes, int n_in,
                              void* d_out, int out_size, void* d_ws, size_t ws_size,
                              hipStream_t stream) {
    (void)n_in; (void)out_size; (void)ws_size;
    const float* x0[4] = {(const float*)d_in[0], (const float*)d_in[1],
                          (const float*)d_in[2], (const float*)d_in[3]};
    const int* eptr[7]; int Ecnt[7];
    for (int r = 0; r < 7; ++r) { eptr[r] = (const int*)d_in[4 + r]; Ecnt[r] = in_sizes[4 + r] / 2; }

    // RELS compute order: ne, en, nt, tn, nk, kn, ee ; node types: 0 news,1 ent,2 topic,3 kg
    static const int relSrc[7]   = {0, 1, 0, 2, 0, 3, 1};
    static const int relDst[7]   = {1, 0, 2, 0, 3, 0, 1};
    // jax flattens dicts sorted by key: ee,en,kn,ne,nk,nt,tn -> position of each compute-order rel
    static const int alphaPos[7] = {3, 1, 5, 6, 4, 2, 0};
    static const int NN[4]       = {100000, 200000, 500, 200000};
    static const int FMAX[4]     = {768, 256, 256, 256};

    const float *Wl[2][7], *Wr[2][7], *Av[2][7], *Bv[2][7];
    const int pbase = 11;  // params start after 4 x's + 7 edge arrays
    for (int L = 0; L < 2; ++L)
        for (int r = 0; r < 7; ++r) {
            int i = pbase + L * 28 + alphaPos[r] * 4;  // per-rel: Wl, Wr, att, b
            Wl[L][r] = (const float*)d_in[i];
            Wr[L][r] = (const float*)d_in[i + 1];
            Av[L][r] = (const float*)d_in[i + 2];
            Bv[L][r] = (const float*)d_in[i + 3];
        }
    const float* lin_b = (const float*)d_in[pbase + 56];
    const float* lin_w = (const float*)d_in[pbase + 57];

    // workspace carve (units of float = 4B)
    float* ws = (float*)d_ws;
    size_t off = 0;
    auto take = [&](size_t n) { float* p = ws + off; off += n; return p; };
    float* y[4]; float* z[4];
    for (int t = 0; t < 4; ++t) y[t] = take((size_t)NN[t] * 256);
    for (int t = 0; t < 4; ++t) z[t] = take((size_t)NN[t] * 128);
    float* xl       = take((size_t)200000 * 256);
    float* xr       = take((size_t)200000 * 256);
    float* ebuf     = take(700000);
    unsigned* menc  = (unsigned*)take(200000);
    float* den      = take(200000);
    unsigned short* xb[4];
    for (int t = 0; t < 4; ++t)
        xb[t] = (unsigned short*)take(((size_t)NN[t] * FMAX[t] + 1) / 2);
    unsigned short* wlb = (unsigned short*)take(98304);  // up to 768*256 bf16 (packed)
    unsigned short* wrb = (unsigned short*)take(98304);

    const float* cur[4] = {x0[0], x0[1], x0[2], x0[3]};
    int curF[4] = {768, 256, 128, 256};

    for (int L = 0; L < 2; ++L) {
        int H = (L == 0) ? 256 : 128;
        float* outb[4];
        for (int t = 0; t < 4; ++t) outb[t] = (L == 0) ? y[t] : z[t];
        // convert this layer's activations to bf16 once
        for (int t = 0; t < 4; ++t) {
            long long n = (long long)NN[t] * curF[t];
            cvt_bf16<<<(unsigned)((n + 255) / 256), 256, 0, stream>>>(cur[t], xb[t], n);
        }
        for (int t = 0; t < 4; ++t) {
            long long n = (long long)NN[t] * H;
            fill_f32<<<(unsigned)((n + 255) / 256), 256, 0, stream>>>(outb[t], 0.f, n);
        }
        for (int r = 0; r < 7; ++r) {
            int s = relSrc[r], d = relDst[r];
            int Ns = NN[s], Nd = NN[d];
            int nwl = curF[s] * H, nwr = curF[d] * H;
            pack_b_bf16<<<ceildiv(nwl, 256), 256, 0, stream>>>(Wl[L][r], wlb, curF[s], H);
            pack_b_bf16<<<ceildiv(nwr, 256), 256, 0, stream>>>(Wr[L][r], wrb, curF[d], H);
            dim3 g1(ceildiv(Ns, BM), H / BN);
            gemm_wmma<<<g1, 256, 0, stream>>>(xb[s], wlb, xl, Ns, H, curF[s]);
            dim3 g2(ceildiv(Nd, BM), H / BN);
            gemm_wmma<<<g2, 256, 0, stream>>>(xb[d], wrb, xr, Nd, H, curF[d]);
            fill_u32<<<ceildiv(Nd, 256), 256, 0, stream>>>(menc, 0x007FFFFFu, Nd); // enc(-inf)
            fill_f32<<<ceildiv(Nd, 256), 256, 0, stream>>>(den, 0.f, Nd);
            int nsl  = (Ns < Nd) ? Ns : Nd;   // PyG self-loops
            int Etot = Ecnt[r] + nsl;
            edge_scores<<<ceildiv(Etot, 8), 256, 0, stream>>>(
                xl, xr, Av[L][r], eptr[r], eptr[r] + Ecnt[r], Ecnt[r], nsl, H, ebuf, menc);
            edge_softmax<<<ceildiv(Etot, 256), 256, 0, stream>>>(
                eptr[r] + Ecnt[r], Ecnt[r], nsl, ebuf, menc, den);
            edge_aggregate<<<ceildiv(Etot, 8), 256, 0, stream>>>(
                xl, ebuf, eptr[r], eptr[r] + Ecnt[r], Ecnt[r], nsl, H, den, outb[d]);
        }
        // biases summed per destination type, then ReLU
        const float* nb[4][3] = {
            {Bv[L][1], Bv[L][3], Bv[L][5]},   // news <- en, tn, kn
            {Bv[L][0], Bv[L][6], nullptr},    // entities <- ne, ee
            {Bv[L][2], nullptr,  nullptr},    // topic <- nt
            {Bv[L][4], nullptr,  nullptr}};   // kg <- nk
        for (int t = 0; t < 4; ++t) {
            long long n = (long long)NN[t] * H;
            relu_bias<<<(unsigned)((n + 255) / 256), 256, 0, stream>>>(
                outb[t], n, H, nb[t][0], nb[t][1], nb[t][2]);
        }
        for (int t = 0; t < 4; ++t) { cur[t] = outb[t]; curF[t] = H; }
    }
    final_linear<<<ceildiv(100000, 256), 256, 0, stream>>>(
        cur[0], lin_w, lin_b, (float*)d_out, 100000, 128);
}
